// HyperLinear_59176059404504
// MI455X (gfx1250) — compile-verified, matrix operand order checked by reading, not run on hardware
//
#include <hip/hip_runtime.h>
#include <hip/hip_bf16.h>

// CDNA5 / gfx1250: wave32, WMMA bf16 16x16x32, fp32 accumulate.
typedef __attribute__((ext_vector_type(16))) __bf16 v16bf;
typedef __attribute__((ext_vector_type(8)))  float  v8f;

#define NOISE_DIM 128
#define IN_F      512
#define OUT_F     512
#define BATCH     256
#define XSTRIDE   (NOISE_DIM + IN_F)     // 640 floats per x row
#define WCOLS     (IN_F * OUT_F + OUT_F) // 262656 columns of W
#define NGROUP    2                      // n-values fused per block
#define NSLABS    65                     // ceil(129 / NGROUP) partial slabs

// Pack two fp32 into bf16x2 dword (lo -> [15:0], hi -> [31:16]).
// Inline asm forces the exact operand pairing (the gfx1250 backend emits this
// mnemonic itself, but SLP mis-pairs the union-cast fallback and then spends
// ~12 VALU repacking halves).
__device__ __forceinline__ unsigned pack_bf16x2(float lo, float hi) {
    unsigned r;
    asm("v_cvt_pk_bf16_f32 %0, %1, %2" : "=v"(r) : "v"(lo), "v"(hi));
    return r;
}

union AFrag { v16bf v; uint4 q[2]; };
union BFrag { v16bf v; uint4 q[2]; };

// ---------------------------------------------------------------------------
// Kernel 0: (a) feats (256x512 fp32, strided in x) -> contiguous bf16;
//           (b) noise (x[:, :128]) -> transposed fp32 noiseT[n][b].
// Pure bandwidth (768 KB); removes all A-side conversion and makes the
// stage-1 noise epilogue a pair of b128 loads per tile.
// ---------------------------------------------------------------------------
__global__ __launch_bounds__(256) void hyper_prep(
    const float* __restrict__ x, unsigned short* __restrict__ fb,
    float* __restrict__ noiseT)
{
    const int t = blockIdx.x * 256 + threadIdx.x;     // 0..32767

    { // feats -> bf16, 4 elements per thread
        const int base = t * 4;
        const int b = base >> 9;
        const int i = base & 511;
        const float4 v = *(const float4*)(x + (size_t)b * XSTRIDE + NOISE_DIM + i);
        uint2 p;
        p.x = pack_bf16x2(v.x, v.y);
        p.y = pack_bf16x2(v.z, v.w);
        *(uint2*)(fb + base) = p;
    }
    { // noise transpose, 1 element per thread (coalesced writes)
        const int n = t >> 8;
        const int b = t & 255;
        noiseT[t] = x[(size_t)b * XSTRIDE + n];
    }
}

// ---------------------------------------------------------------------------
// Stage 1: block = (o-tile of 64) x (group of NGROUP n-values).
// For each n: G_n[b,o] = sum_{i<=512} F[b,i]*Wa[n, i*512+o]  via bf16 WMMA
// (F[b,512]==1 folded as a direct add of row i=512), then fused epilogue
//   accOut += noise[b,n] * G_n     (noise[b,128] == 1 for the appended b row)
// so only NSLABS noise-weighted fp32 partials hit memory.
// 256 threads = 8 waves. Wave grid 8(M) x 1(N): each wave owns 32 rows x 64
// cols = 2x4 WMMA tiles -> A fragments loaded once per wave (no duplication),
// B reuse served from LDS. K loop: 16 chunks of 32, double-buffered bf16 W
// tile in LDS (transposed [o][k]).
// ---------------------------------------------------------------------------
__global__ __launch_bounds__(256) void hyper_stage1_wmma(
    const float* __restrict__ W, const float* __restrict__ bvec,
    const unsigned short* __restrict__ fb, const float* __restrict__ noiseT,
    float* __restrict__ G)
{
    __shared__ __align__(16) unsigned short ldsW[2][64 * 32]; // 2 x 4KB ping-pong

    const int ob = blockIdx.x * 64;       // output-column block base

    const int tid   = threadIdx.x;
    const int lane  = tid & 31;
    const int wave  = tid >> 5;           // M octant: rows wave*32 .. +31
    const int lhalf = lane >> 4;          // 0/1 (K-half select in frags)
    const int l16   = lane & 15;

    // staging: 256 threads cover 32(k) x 64(o); each thread 2(k) x 4(o) so
    // k-adjacent bf16 pairs pack into single ds_store_b32.
    const int kp = (tid >> 4) * 2;        // k row pair base, 0,2,..,30
    const int cg = (tid & 15) * 4;        // column group base, 0..60

    v8f accOut[2][4];
#pragma unroll
    for (int i = 0; i < 2; ++i)
#pragma unroll
        for (int j = 0; j < 4; ++j)
#pragma unroll
            for (int e = 0; e < 8; ++e) accOut[i][j][e] = 0.0f;

#pragma unroll
    for (int ni = 0; ni < NGROUP; ++ni) {
        const int n = blockIdx.y * NGROUP + ni;
        if (n > NOISE_DIM) break;         // last group holds only n == 128
        const float* __restrict__ Wn =
            (n < NOISE_DIM) ? (W + (size_t)n * WCOLS) : bvec;

        v8f acc[2][4];
#pragma unroll
        for (int i = 0; i < 2; ++i)
#pragma unroll
            for (int j = 0; j < 4; ++j)
#pragma unroll
                for (int e = 0; e < 8; ++e) acc[i][j][e] = 0.0f;

        // prologue: stage chunk 0 into buffer 0
        {
            const float* r0 = Wn + (size_t)kp * OUT_F + ob + cg;
            const float4 a = *(const float4*)r0;
            const float4 b = *(const float4*)(r0 + OUT_F);
            const float la[4] = {a.x, a.y, a.z, a.w};
            const float lb[4] = {b.x, b.y, b.z, b.w};
#pragma unroll
            for (int j = 0; j < 4; ++j)
                *(unsigned*)&ldsW[0][(cg + j) * 32 + kp] = pack_bf16x2(la[j], lb[j]);
        }
        __syncthreads();

        for (int kc = 0; kc < 16; ++kc) {
            const int cur = kc & 1;

            // ---- stage NEXT chunk into other buffer (overlaps with WMMA) ---
            if (kc + 1 < 16) {
                const int k1 = (kc + 1) * 32;
                const float* r0 = Wn + (size_t)(k1 + kp) * OUT_F + ob + cg;
                const float4 a = *(const float4*)r0;
                const float4 b = *(const float4*)(r0 + OUT_F);
                const float la[4] = {a.x, a.y, a.z, a.w};
                const float lb[4] = {b.x, b.y, b.z, b.w};
#pragma unroll
                for (int j = 0; j < 4; ++j)
                    *(unsigned*)&ldsW[cur ^ 1][(cg + j) * 32 + kp] =
                        pack_bf16x2(la[j], lb[j]);
                if (kc + 2 < 16) // global_prefetch_b8 two chunks ahead
                    __builtin_prefetch(
                        Wn + (size_t)(k1 + 32 + kp) * OUT_F + ob + cg, 0, 1);
            }

            // ---- A fragments: bf16 feats, straight b128 loads --------------
            const int k0 = kc * 32;
            AFrag af[2];
#pragma unroll
            for (int mt = 0; mt < 2; ++mt) {
                const int m = wave * 32 + mt * 16 + l16;
                const unsigned short* fr = fb + (size_t)m * IN_F + k0 + lhalf * 8;
                af[mt].q[0] = *(const uint4*)fr;
                af[mt].q[1] = *(const uint4*)(fr + 16);
            }

            // ---- B fragments: column-contiguous ds_load_b128 ---------------
            BFrag bfr[4];
#pragma unroll
            for (int nt = 0; nt < 4; ++nt) {
                const int oc = nt * 16 + l16;
                bfr[nt].q[0] = *(const uint4*)&ldsW[cur][oc * 32 + lhalf * 8];
                bfr[nt].q[1] = *(const uint4*)&ldsW[cur][oc * 32 + lhalf * 8 + 16];
            }

            // ---- 8 WMMAs per K chunk ---------------------------------------
#pragma unroll
            for (int mt = 0; mt < 2; ++mt)
#pragma unroll
                for (int nt = 0; nt < 4; ++nt)
                    acc[mt][nt] = __builtin_amdgcn_wmma_f32_16x16x32_bf16(
                        false, af[mt].v, false, bfr[nt].v,
                        (short)0, acc[mt][nt], false, false);

            __syncthreads();
        }

        // ---- F[b,512]==1 : add Wa[n, 512*512 + o] into every C row ---------
        // C layout: lane -> N = lane%16, VGPR v -> M = v + 8*(lane/16).
#pragma unroll
        for (int nt = 0; nt < 4; ++nt) {
            const int oc = ob + nt * 16 + l16;
            const float bw = Wn[(size_t)IN_F * OUT_F + oc];
#pragma unroll
            for (int mt = 0; mt < 2; ++mt)
#pragma unroll
                for (int v = 0; v < 8; ++v) acc[mt][nt][v] += bw;
        }

        // ---- fused epilogue: accOut += noise[b,n] * acc --------------------
        // Thread's rows per mt are 8 consecutive b: two b128 loads from noiseT.
#pragma unroll
        for (int mt = 0; mt < 2; ++mt) {
            float sv[8];
            if (n < NOISE_DIM) {
                const int m0 = wave * 32 + mt * 16 + lhalf * 8;
                const float4 u0 = *(const float4*)(noiseT + (size_t)n * BATCH + m0);
                const float4 u1 = *(const float4*)(noiseT + (size_t)n * BATCH + m0 + 4);
                sv[0]=u0.x; sv[1]=u0.y; sv[2]=u0.z; sv[3]=u0.w;
                sv[4]=u1.x; sv[5]=u1.y; sv[6]=u1.z; sv[7]=u1.w;
            } else {
#pragma unroll
                for (int v = 0; v < 8; ++v) sv[v] = 1.0f;  // appended A column
            }
#pragma unroll
            for (int nt = 0; nt < 4; ++nt)
#pragma unroll
                for (int v = 0; v < 8; ++v)
                    accOut[mt][nt][v] = fmaf(sv[v], acc[mt][nt][v], accOut[mt][nt][v]);
        }
    }

    // ---- store weighted partial slab P[g][m][o] ----------------------------
    float* __restrict__ Gn = G + (size_t)blockIdx.y * (BATCH * OUT_F);
#pragma unroll
    for (int mt = 0; mt < 2; ++mt)
#pragma unroll
        for (int v = 0; v < 8; ++v) {
            const int m = wave * 32 + mt * 16 + lhalf * 8 + v;
#pragma unroll
            for (int nt = 0; nt < 4; ++nt) {
                const int oc = ob + nt * 16 + l16;
                Gn[(size_t)m * OUT_F + oc] = accOut[mt][nt][v];
            }
        }
}

// ---------------------------------------------------------------------------
// Stage 2: out[b,o] = sum_{g < NSLABS} P[g,b,o].
// Noise weights already folded in -> plain deterministic slab sum, float4/thread.
// ---------------------------------------------------------------------------
__global__ __launch_bounds__(256) void hyper_stage2_reduce(
    const float* __restrict__ G, float* __restrict__ out)
{
    const int idx4 = (blockIdx.x * 256 + threadIdx.x) * 4;
    const size_t stride = (size_t)BATCH * OUT_F;

    float4 s = *(const float4*)(G + idx4);
#pragma unroll 8
    for (int g = 1; g < NSLABS; ++g) {
        const float4 p = *(const float4*)(G + (size_t)g * stride + idx4);
        s.x += p.x; s.y += p.y; s.z += p.z; s.w += p.w;
    }
    *(float4*)(out + idx4) = s;
}

extern "C" void kernel_launch(void* const* d_in, const int* in_sizes, int n_in,
                              void* d_out, int out_size, void* d_ws, size_t ws_size,
                              hipStream_t stream) {
    const float* x    = (const float*)d_in[0];  // (256, 640) fp32
    const float* W    = (const float*)d_in[1];  // (128, 262656) fp32
    const float* bvec = (const float*)d_in[2];  // (262656,) fp32
    float* out = (float*)d_out;                 // (256, 512) fp32

    // ws layout: [0,256KB)        feats bf16 (256x512)
    //            [256KB,384KB)    noiseT fp32 (128x256)
    //            [384KB,~34.4MB)  P partial slabs fp32 (65 x 256 x 512)
    char* wsb = (char*)d_ws;
    unsigned short* fb = (unsigned short*)wsb;
    float* noiseT = (float*)(wsb + (size_t)BATCH * IN_F * sizeof(unsigned short));
    float* G      = (float*)(wsb + (size_t)BATCH * IN_F * sizeof(unsigned short)
                                 + (size_t)NOISE_DIM * BATCH * sizeof(float));

    (void)in_sizes; (void)n_in; (void)out_size; (void)ws_size;

    hyper_prep<<<dim3(128), dim3(256), 0, stream>>>(x, fb, noiseT);

    dim3 g1(OUT_F / 64, NSLABS);                // 8 x 65 = 520 workgroups
    hyper_stage1_wmma<<<g1, dim3(256), 0, stream>>>(W, bvec, fb, noiseT, G);

    dim3 g2((BATCH * OUT_F) / (256 * 4));       // 128 workgroups
    hyper_stage2_reduce<<<g2, dim3(256), 0, stream>>>(G, out);
}
